// OuroborosAttention_68685116998082
// MI455X (gfx1250) — matrix operand order repaired and checked by reading, not compiled, not run on hardware
//
#include <hip/hip_runtime.h>
#include <hip/hip_bf16.h>

typedef __bf16 bf16;
typedef __attribute__((ext_vector_type(8)))  __bf16 bv8;
typedef __attribute__((ext_vector_type(16))) __bf16 bv16;
typedef __attribute__((ext_vector_type(8)))  float  v8f;
typedef __attribute__((ext_vector_type(4)))  unsigned u32x4;
typedef __attribute__((ext_vector_type(8)))  int      i32x8;
typedef __attribute__((ext_vector_type(4)))  int      i32x4;

#define B_  4
#define T_  2048
#define D_  2048
#define H_  16
#define HD_ 128
#define E3_ 6144
#define BT_ 8192

#if __has_builtin(__builtin_amdgcn_global_load_async_to_lds_b128)
#define HAS_ASYNC_LDS 1
#else
#define HAS_ASYNC_LDS 0
#endif

#if __has_builtin(__builtin_amdgcn_tensor_load_to_lds)
#define HAS_TDM 1
#else
#define HAS_TDM 0
#endif

// ---------------------------------------------------------------------------
// WMMA helper (CDNA5 v_wmma_f32_16x16x32_bf16)
// ---------------------------------------------------------------------------
__device__ __forceinline__ v8f wmma_bf16(bv16 a, bv16 b, v8f c) {
  return __builtin_amdgcn_wmma_f32_16x16x32_bf16(
      false, a, false, b, (short)0, c, false, false);
}

// A-matrix 16x32 bf16 fragment from row-major (M,K) storage.
// lanes 0-15: M=0..15; K in {h8..h8+7, 16+h8..16+h8+7}, h8 = 8*(lane>>4)
__device__ __forceinline__ bv16 load_frag_a(const bf16* p, int ld) {
  int lane = threadIdx.x & 31;
  const bf16* r = p + (size_t)(lane & 15) * ld + ((lane >> 4) << 3);
  bv8 lo = *(const bv8*)(r);
  bv8 hi = *(const bv8*)(r + 16);
  bv16 f;
#pragma unroll
  for (int i = 0; i < 8; ++i) { f[i] = lo[i]; f[i + 8] = hi[i]; }
  return f;
}

// B-matrix 32x16 bf16 fragment from (N,K)-row-major storage (row index = N).
// lane holds N=lane&15; K in {h8..h8+7, 16+h8..16+h8+7}, h8 = 8*(lane>>4)
__device__ __forceinline__ bv16 load_frag_b(const bf16* p, int ld) {
  int lane = threadIdx.x & 31;
  const bf16* r = p + (size_t)(lane & 15) * ld + ((lane >> 4) << 3);
  bv8 lo = *(const bv8*)(r);
  bv8 hi = *(const bv8*)(r + 16);
  bv16 f;
#pragma unroll
  for (int i = 0; i < 8; ++i) { f[i] = lo[i]; f[i + 8] = hi[i]; }
  return f;
}

__device__ __forceinline__ void wait_dscnt0() {
#if __has_builtin(__builtin_amdgcn_s_wait_dscnt)
  __builtin_amdgcn_s_wait_dscnt(0);
#else
  asm volatile("s_wait_dscnt 0" ::: "memory");
#endif
}

template <int N>
__device__ __forceinline__ void wait_asynccnt() {
#if HAS_ASYNC_LDS
#if __has_builtin(__builtin_amdgcn_s_wait_asynccnt)
  __builtin_amdgcn_s_wait_asynccnt(N);
#else
  asm volatile("s_wait_asynccnt %0" ::"i"(N) : "memory");
#endif
#endif
}

// 16-byte global -> LDS copy: async DMA when available, sync fallback.
__device__ __forceinline__ void copy_b128_to_lds(const bf16* g, bf16* l) {
#if HAS_ASYNC_LDS
  typedef __attribute__((address_space(1))) i32x4 gv4i;
  typedef __attribute__((address_space(3))) i32x4 lv4i;
  __builtin_amdgcn_global_load_async_to_lds_b128(
      (gv4i*)(uintptr_t)g,
      (lv4i*)(unsigned)(uintptr_t)l,
      0, 0);
#else
  *(bv8*)l = *(const bv8*)g;
#endif
}

// ---------------------------------------------------------------------------
// Tensor Data Mover: 2-D tile (tile_x cols x tile_y rows, 2-byte elems)
// from row-major tensor (stride_x elems per row) into LDS.
// ---------------------------------------------------------------------------
#if HAS_TDM
__device__ __forceinline__ void tdm_load_2d(const void* gaddr, unsigned lds_off,
                                            int tile_x, int tile_y,
                                            int tensor_x, int tensor_y,
                                            int stride_x) {
  unsigned long long ga = (unsigned long long)(uintptr_t)gaddr;
  u32x4 g0;
  g0[0] = 1u;                                           // count=1 (valid D#)
  g0[1] = lds_off;                                      // lds_addr
  g0[2] = (unsigned)(ga & 0xFFFFFFFFu);                 // global_addr[31:0]
  g0[3] = (unsigned)((ga >> 32) & 0x1FFFFFFu) | (2u << 30); // [56:32] | type=2
  i32x8 g1;
  g1[0] = 1 << 16;                                      // data_size = 2 bytes
  g1[1] = (tensor_x & 0xFFFF) << 16;                    // tensor_dim0[15:0]
  g1[2] = ((tensor_x >> 16) & 0xFFFF) | ((tensor_y & 0xFFFF) << 16);
  g1[3] = ((tensor_y >> 16) & 0xFFFF) | ((tile_x & 0xFFFF) << 16);
  g1[4] = tile_y & 0xFFFF;                              // tile_dim1 (tile_dim2=0)
  g1[5] = stride_x;                                     // tensor_dim0_stride lo
  g1[6] = 0;
  g1[7] = 0;
  i32x4 z4 = {0, 0, 0, 0};
#if __has_include(<hip/amd_detail/amd_gfx1250_TDM.h>)
  i32x8 z8 = {0, 0, 0, 0, 0, 0, 0, 0};
  __builtin_amdgcn_tensor_load_to_lds(g0, g1, z4, z4, z8, 0);   // 6-arg toolchain
#else
  __builtin_amdgcn_tensor_load_to_lds(g0, g1, z4, z4, 0);       // 5-arg toolchain
#endif
}

__device__ __forceinline__ void wait_tensorcnt0() {
#if __has_builtin(__builtin_amdgcn_s_wait_tensorcnt)
  __builtin_amdgcn_s_wait_tensorcnt(0);
#else
  asm volatile("s_wait_tensorcnt 0" ::: "memory");
#endif
}
#endif

// ---------------------------------------------------------------------------
// fp32 -> bf16 conversion (weights)
// ---------------------------------------------------------------------------
__global__ __launch_bounds__(256) void cvt_bf16_kernel(const float* __restrict__ s,
                                                       bf16* __restrict__ d, int n) {
  int i = blockIdx.x * 256 + threadIdx.x;
  int stride = gridDim.x * 256;
  for (; i < n; i += stride) d[i] = (bf16)s[i];
}

// ---------------------------------------------------------------------------
// RMSNorm: one 256-thread block per row of 2048
// ---------------------------------------------------------------------------
__global__ __launch_bounds__(256) void rmsnorm_kernel(const float* __restrict__ x,
                                                      const float* __restrict__ w,
                                                      bf16* __restrict__ out) {
  __shared__ float red[8];
  int row = blockIdx.x;
  const float* xr = x + (size_t)row * D_;
  float ss = 0.f;
  for (int i = threadIdx.x; i < D_; i += 256) { float v = xr[i]; ss += v * v; }
#pragma unroll
  for (int off = 16; off; off >>= 1) ss += __shfl_xor(ss, off, 32);
  if ((threadIdx.x & 31) == 0) red[threadIdx.x >> 5] = ss;
  __syncthreads();
  float tot = 0.f;
#pragma unroll
  for (int i = 0; i < 8; ++i) tot += red[i];
  float inv = rsqrtf(tot * (1.0f / D_) + 1e-5f);
  bf16* orow = out + (size_t)row * D_;
  for (int i = threadIdx.x; i < D_; i += 256) orow[i] = (bf16)(xr[i] * inv * w[i]);
}

// ---------------------------------------------------------------------------
// GEMM core: block tile 128x128, K-step 32, double-buffered async LDS staging.
// Block = 256 threads = 8 waves in 4(M) x 2(N); wave tile = 32x64.
// ---------------------------------------------------------------------------
__device__ __forceinline__ void stage128x32(const bf16* g, int ld, bf16* l) {
  int tid = threadIdx.x;
#pragma unroll
  for (int rep = 0; rep < 2; ++rep) {
    int lin = (rep * 256 + tid) * 8;          // 4096 bf16 total
    int r = lin >> 5, c = lin & 31;
    if (rep == 0) __builtin_prefetch(g + (size_t)r * ld + c + 64, 0, 1);
    copy_b128_to_lds(g + (size_t)r * ld + c, l + lin);
  }
}

__device__ __forceinline__ void gemm_tiles(const bf16* __restrict__ A,
                                           const bf16* __restrict__ W,
                                           int Kdim, v8f acc[2][4]) {
  __shared__ __align__(16) bf16 sA[2][128 * 32];
  __shared__ __align__(16) bf16 sB[2][128 * 32];
  int mblock = blockIdx.y * 128;
  int nblock = blockIdx.x * 128;
  int w = threadIdx.x >> 5;
  int wm = w & 3, wn = w >> 2;

  const int NK = Kdim / 32;
  stage128x32(A + (size_t)mblock * Kdim, Kdim, sA[0]);
  stage128x32(W + (size_t)nblock * Kdim, Kdim, sB[0]);
  for (int kb = 0; kb < NK; ++kb) {
    int cur = kb & 1;
    if (kb + 1 < NK) {
      stage128x32(A + (size_t)mblock * Kdim + (kb + 1) * 32, Kdim, sA[cur ^ 1]);
      stage128x32(W + (size_t)nblock * Kdim + (kb + 1) * 32, Kdim, sB[cur ^ 1]);
      wait_asynccnt<4>();                   // in-order: current buffer done
    } else {
      wait_asynccnt<0>();
    }
    __syncthreads();

    const bf16* pA = &sA[cur][(wm * 32) * 32];
    bv16 a0 = load_frag_a(pA, 32);
    bv16 a1 = load_frag_a(pA + 16 * 32, 32);
#pragma unroll
    for (int j = 0; j < 4; ++j) {
      bv16 b = load_frag_b(&sB[cur][(wn * 64 + 16 * j) * 32], 32);
      acc[0][j] = wmma_bf16(a0, b, acc[0][j]);
      acc[1][j] = wmma_bf16(a1, b, acc[1][j]);
    }
    __syncthreads();
  }
}

// QKV projection: h(8192x2048) x Wqkv(6144x2048)^T -> bf16 Q/K/V [B,H,T,hd]
__global__ __launch_bounds__(256) void gemm_qkv_kernel(const bf16* __restrict__ A,
                                                       const bf16* __restrict__ W,
                                                       bf16* __restrict__ Qb,
                                                       bf16* __restrict__ Kb,
                                                       bf16* __restrict__ Vb) {
  int w = threadIdx.x >> 5;
  int wm = w & 3, wn = w >> 2;
  int mbase = blockIdx.y * 128 + wm * 32;
  int nbase = blockIdx.x * 128 + wn * 64;
  v8f acc[2][4] = {};
  gemm_tiles(A, W, D_, acc);

  int lane = threadIdx.x & 31;
  int nloc = lane & 15;
  int mhalf = (lane >> 4) << 3;
#pragma unroll
  for (int i = 0; i < 2; ++i) {
#pragma unroll
    for (int j = 0; j < 4; ++j) {
      int ntile = nbase + 16 * j;            // tile never straddles 2048
      int which = ntile >> 11;
      bf16* dp = (which == 0) ? Qb : ((which == 1) ? Kb : Vb);
      int rem = (ntile & 2047) + nloc;
      int h = rem >> 7, d = rem & 127;
#pragma unroll
      for (int r = 0; r < 8; ++r) {
        int m = mbase + 16 * i + mhalf + r;
        int b = m >> 11, t = m & 2047;
        size_t dst = (((size_t)(b * H_ + h) * T_) + t) * HD_ + d;
        dp[dst] = (bf16)acc[i][j][r];
      }
    }
  }
}

// Output projection: attn(8192x2048) x wo(2048x2048)^T -> fp32 out
__global__ __launch_bounds__(256) void gemm_out_kernel(const bf16* __restrict__ A,
                                                       const bf16* __restrict__ W,
                                                       float* __restrict__ out) {
  int w = threadIdx.x >> 5;
  int wm = w & 3, wn = w >> 2;
  int mbase = blockIdx.y * 128 + wm * 32;
  int nbase = blockIdx.x * 128 + wn * 64;
  v8f acc[2][4] = {};
  gemm_tiles(A, W, D_, acc);

  int lane = threadIdx.x & 31;
  int nloc = lane & 15;
  int mhalf = (lane >> 4) << 3;
#pragma unroll
  for (int i = 0; i < 2; ++i)
#pragma unroll
    for (int j = 0; j < 4; ++j)
#pragma unroll
      for (int r = 0; r < 8; ++r) {
        int m = mbase + 16 * i + mhalf + r;
        int n = nbase + 16 * j + nloc;
        out[(size_t)m * D_ + n] = acc[i][j][r];
      }
}

// ---------------------------------------------------------------------------
// RoPE on Q and K in place ([B,H,T,hd] bf16); one thread per (b,h,t,pair)
// ---------------------------------------------------------------------------
__global__ __launch_bounds__(256) void rope_kernel(bf16* __restrict__ Qb,
                                                   bf16* __restrict__ Kb,
                                                   const float* __restrict__ fc,
                                                   const float* __restrict__ fs) {
  int idx = blockIdx.x * 256 + threadIdx.x;   // B*H*T*64 threads
  int p = idx & 63;
  int t = (idx >> 6) & 2047;
  int bh = idx >> 17;
  size_t base = (((size_t)bh * T_) + t) * HD_ + 2 * p;
  float c = fc[t * 64 + p], s = fs[t * 64 + p];
  float qr = (float)Qb[base], qi = (float)Qb[base + 1];
  Qb[base]     = (bf16)(qr * c - qi * s);
  Qb[base + 1] = (bf16)(qr * s + qi * c);
  float kr = (float)Kb[base], ki = (float)Kb[base + 1];
  Kb[base]     = (bf16)(kr * c - ki * s);
  Kb[base + 1] = (bf16)(kr * s + ki * c);
}

// ---------------------------------------------------------------------------
// Causal flash attention. Block = 8 waves, 128 Q rows (16/wave), one (b,h).
// K/V 32x128 tiles staged via TDM (or async copies); V transposed in LDS;
// P round-trips through per-wave LDS (C-layout -> A-layout).
// ---------------------------------------------------------------------------
__global__ __launch_bounds__(256) void attn_kernel(const bf16* __restrict__ Qg,
                                                   const bf16* __restrict__ Kg,
                                                   const bf16* __restrict__ Vg,
                                                   bf16* __restrict__ Oa) {
  __shared__ __align__(16) bf16 sK[32 * 128];
  __shared__ __align__(16) bf16 sV[32 * 128];
  __shared__ __align__(16) bf16 sVt[128 * 32];
  __shared__ __align__(16) bf16 sP[8][16 * 32];

  int bh = blockIdx.y;
  int qblk = blockIdx.x;
  int w = threadIdx.x >> 5;
  int lane = threadIdx.x & 31;
  int nloc = lane & 15;
  int mhalf = (lane >> 4) << 3;

  const bf16* Qh = Qg + (size_t)bh * T_ * HD_;
  const bf16* Kh = Kg + (size_t)bh * T_ * HD_;
  const bf16* Vh = Vg + (size_t)bh * T_ * HD_;

  int qbase = qblk * 128 + w * 16;

  bv16 qf[4];
#pragma unroll
  for (int kk = 0; kk < 4; ++kk)
    qf[kk] = load_frag_a(Qh + (size_t)qbase * HD_ + kk * 32, HD_);

  v8f o[8] = {};
  float rmax[8], rsum[8];
#pragma unroll
  for (int r = 0; r < 8; ++r) { rmax[r] = -1e30f; rsum[r] = 0.f; }
  const float scale = 0.08838834764831845f;   // 1/sqrt(128)

  int nkb = (qblk + 1) * 4;                   // causal
  for (int kb = 0; kb < nkb; ++kb) {
    int k0 = kb * 32;
    // ---- stage K and V rows (32x128 each) ----
#if HAS_TDM
    if (w == 0) {
      tdm_load_2d(Kh + (size_t)k0 * HD_, (unsigned)(uintptr_t)&sK[0],
                  HD_, 32, HD_, T_ - k0, HD_);
      tdm_load_2d(Vh + (size_t)k0 * HD_, (unsigned)(uintptr_t)&sV[0],
                  HD_, 32, HD_, T_ - k0, HD_);
      wait_tensorcnt0();
    }
#else
#pragma unroll
    for (int rep = 0; rep < 2; ++rep) {
      int lin = (rep * 256 + threadIdx.x) * 8;
      copy_b128_to_lds(&Kh[(size_t)k0 * HD_ + lin], &sK[lin]);
      copy_b128_to_lds(&Vh[(size_t)k0 * HD_ + lin], &sV[lin]);
    }
    wait_asynccnt<0>();
#endif
    __syncthreads();
    // ---- transpose V in LDS: sV(32x128) -> sVt(128x32) ----
#pragma unroll
    for (int rep = 0; rep < 16; ++rep) {
      int lin = rep * 256 + threadIdx.x;
      int vr = lin >> 7, vc = lin & 127;
      sVt[vc * 32 + vr] = sV[vr * 128 + vc];
    }
    __syncthreads();

    // ---- S = Q K^T (two 16x16 tiles over this 32-key block) ----
    v8f s[2] = {};
#pragma unroll
    for (int j = 0; j < 2; ++j)
#pragma unroll
      for (int kk = 0; kk < 4; ++kk)
        s[j] = wmma_bf16(qf[kk], load_frag_b(&sK[(j * 16) * 128 + kk * 32], 128), s[j]);

    // ---- online softmax; write P (bf16) into per-wave LDS ----
#pragma unroll
    for (int r = 0; r < 8; ++r) {
      int qrow = qbase + mhalf + r;
      float s0 = s[0][r] * scale;
      float s1 = s[1][r] * scale;
      if (k0 + nloc      > qrow) s0 = -1e30f;
      if (k0 + 16 + nloc > qrow) s1 = -1e30f;
      float mx = fmaxf(s0, s1);
      mx = fmaxf(mx, __shfl_xor(mx, 1, 32));
      mx = fmaxf(mx, __shfl_xor(mx, 2, 32));
      mx = fmaxf(mx, __shfl_xor(mx, 4, 32));
      mx = fmaxf(mx, __shfl_xor(mx, 8, 32));
      float mnew = fmaxf(rmax[r], mx);
      float corr = __expf(rmax[r] - mnew);
      float p0 = __expf(s0 - mnew);
      float p1 = __expf(s1 - mnew);
      float ps = p0 + p1;
      ps += __shfl_xor(ps, 1, 32);
      ps += __shfl_xor(ps, 2, 32);
      ps += __shfl_xor(ps, 4, 32);
      ps += __shfl_xor(ps, 8, 32);
      rsum[r] = rsum[r] * corr + ps;
      rmax[r] = mnew;
#pragma unroll
      for (int t = 0; t < 8; ++t) o[t][r] *= corr;
      int prow = mhalf + r;
      sP[w][prow * 32 + nloc]      = (bf16)p0;
      sP[w][prow * 32 + 16 + nloc] = (bf16)p1;
    }
    wait_dscnt0();   // same-wave DS store->load: DS pipe in-order, drain

    // ---- O += P * V ----
    bv16 pf = load_frag_a(&sP[w][0], 32);
#pragma unroll
    for (int t = 0; t < 8; ++t)
      o[t] = wmma_bf16(pf, load_frag_b(&sVt[(t * 16) * 32], 32), o[t]);
    __syncthreads();
  }

  // ---- normalize and write bf16 attn output as (B*T, D) rows ----
  int b = bh >> 4, h = bh & 15;
#pragma unroll
  for (int r = 0; r < 8; ++r) {
    float inv = 1.f / rsum[r];
    int qrow = qbase + mhalf + r;
    size_t rowbase = ((size_t)(b * T_ + qrow)) * D_ + h * HD_;
#pragma unroll
    for (int t = 0; t < 8; ++t)
      Oa[rowbase + t * 16 + nloc] = (bf16)(o[t][r] * inv);
  }
}

// ---------------------------------------------------------------------------
// Launch
// ---------------------------------------------------------------------------
extern "C" void kernel_launch(void* const* d_in, const int* in_sizes, int n_in,
                              void* d_out, int out_size, void* d_ws, size_t ws_size,
                              hipStream_t stream) {
  const float* x    = (const float*)d_in[0];
  const float* fc   = (const float*)d_in[1];
  const float* fs   = (const float*)d_in[2];
  const float* wn   = (const float*)d_in[3];
  const float* Wqkv = (const float*)d_in[4];
  const float* wo   = (const float*)d_in[5];
  float* out = (float*)d_out;

  size_t off = 0;
  char* wsb = (char*)d_ws;
  auto alloc = [&](size_t bytes) -> void* {
    void* p = wsb + off;
    off += (bytes + 255) & ~(size_t)255;
    return p;
  };
  bf16* hbf   = (bf16*)alloc((size_t)BT_ * D_ * 2);
  bf16* wqkvb = (bf16*)alloc((size_t)E3_ * D_ * 2);
  bf16* wob   = (bf16*)alloc((size_t)D_ * D_ * 2);
  bf16* Qb    = (bf16*)alloc((size_t)B_ * H_ * T_ * HD_ * 2);
  bf16* Kb    = (bf16*)alloc((size_t)B_ * H_ * T_ * HD_ * 2);
  bf16* Vb    = (bf16*)alloc((size_t)B_ * H_ * T_ * HD_ * 2);
  bf16* attnb = (bf16*)alloc((size_t)BT_ * D_ * 2);

  cvt_bf16_kernel<<<2048, 256, 0, stream>>>(Wqkv, wqkvb, E3_ * D_);
  cvt_bf16_kernel<<<2048, 256, 0, stream>>>(wo, wob, D_ * D_);
  rmsnorm_kernel<<<BT_, 256, 0, stream>>>(x, wn, hbf);
  gemm_qkv_kernel<<<dim3(E3_ / 128, BT_ / 128), 256, 0, stream>>>(hbf, wqkvb, Qb, Kb, Vb);
  rope_kernel<<<(B_ * H_ * T_ * 64) / 256, 256, 0, stream>>>(Qb, Kb, fc, fs);
  attn_kernel<<<dim3(T_ / 128, B_ * H_), 256, 0, stream>>>(Qb, Kb, Vb, attnb);
  gemm_out_kernel<<<dim3(D_ / 128, BT_ / 128), 256, 0, stream>>>(attnb, wob, out);
}
